// OptimusPrime2_52905407152503
// MI455X (gfx1250) — matrix operand order repaired and checked by reading, not compiled, not run on hardware
//
#include <hip/hip_runtime.h>

// ---------------------------------------------------------------------------
// Problem constants
// ---------------------------------------------------------------------------
#define S_LEN   100
#define BATCH   32
#define NHEAD   4
#define HDIM    128
#define DMODEL  512
#define FLAT_IN 15873
#define FLAT_PAD 15936      // padded to multiple of 64
#define HIN_  143
#define WIN_  111
#define HOUT_ 135
#define WOUT_ 103

// ---------------------------------------------------------------------------
// WMMA types (CDNA5 / gfx1250, wave32)
// ---------------------------------------------------------------------------
typedef __attribute__((ext_vector_type(16))) __bf16 bf16x16;
typedef __attribute__((ext_vector_type(8)))  float  floatx8;

union Frag {
    uint4   u[2];
    bf16x16 v;
};

__device__ __forceinline__ unsigned short f2bf(float f) {
    unsigned int u = __float_as_uint(f);
    u += 0x7fffu + ((u >> 16) & 1u);          // round-to-nearest-even
    return (unsigned short)(u >> 16);
}

// ---------------------------------------------------------------------------
// Tiled bf16 WMMA GEMM:  C[M,N] = act( A[M,K] * W[N,K]^T + bias[N] )
// BM=128, BN=128, BK=32; 256 threads = 8 waves (4x2), each wave 32x64 =
// 2x4 tiles of v_wmma_f32_16x16x32_bf16.  Double-buffered LDS staged with
// CDNA5 async global->LDS DMA (ASYNCcnt) on interior blocks.
// Requires K % 32 == 0; M,N arbitrary (bounds-checked, sync fallback path).
// ---------------------------------------------------------------------------
#define BM 128
#define BN 128
#define BKK 32

// dynamic LDS layout (bytes): A buf0 | A buf1 | B buf0 | B buf1  (8 KB each)
#define A_LOFF(buf) ((buf) * 8192)
#define B_LOFF(buf) (16384 + (buf) * 8192)
#define GEMM_LDS_BYTES 32768

__global__ __launch_bounds__(256)
void gemm_bf16_kernel(const unsigned short* __restrict__ A,
                      const unsigned short* __restrict__ W,
                      const float* __restrict__ bias,
                      float* __restrict__ C,
                      int M, int N, int K, int relu)
{
    extern __shared__ __align__(16) unsigned short sm[];

    const int tid  = threadIdx.x;
    const int lane = tid & 31;
    const int wave = tid >> 5;
    const int wm   = wave >> 1;     // 0..3 -> 32-row strips
    const int wn   = wave & 1;      // 0..1 -> 64-col strips
    const int m0   = blockIdx.y * BM;
    const int n0   = blockIdx.x * BN;
    const bool interior = (m0 + BM <= M) && (n0 + BN <= N);

    floatx8 acc[2][4] = {};

    // staging chunks: tile = 128 rows x 32 cols bf16 = 512 x 16B chunks,
    // 2 chunks per thread per tile (A and B each)
    const int cA0 = tid, cA1 = tid + 256;
    const int r0 = cA0 >> 2, k0e = (cA0 & 3) * 8;
    const int r1 = cA1 >> 2, k1e = (cA1 & 3) * 8;

    // LDS byte offset of start of dynamic shared memory
    const unsigned smbase = __builtin_amdgcn_groupstaticsize();

    auto issueAsync = [&](int kk, int buf) {
        unsigned la0 = smbase + A_LOFF(buf) + (unsigned)(r0 * BKK + k0e) * 2u;
        unsigned la1 = smbase + A_LOFF(buf) + (unsigned)(r1 * BKK + k1e) * 2u;
        unsigned lb0 = smbase + B_LOFF(buf) + (unsigned)(r0 * BKK + k0e) * 2u;
        unsigned lb1 = smbase + B_LOFF(buf) + (unsigned)(r1 * BKK + k1e) * 2u;
        const unsigned short* ga0 = A + (size_t)(m0 + r0) * K + kk + k0e;
        const unsigned short* ga1 = A + (size_t)(m0 + r1) * K + kk + k1e;
        const unsigned short* gb0 = W + (size_t)(n0 + r0) * K + kk + k0e;
        const unsigned short* gb1 = W + (size_t)(n0 + r1) * K + kk + k1e;
        asm volatile(
            "global_load_async_to_lds_b128 %0, %1, off\n\t"
            "global_load_async_to_lds_b128 %2, %3, off\n\t"
            "global_load_async_to_lds_b128 %4, %5, off\n\t"
            "global_load_async_to_lds_b128 %6, %7, off"
            :
            : "v"(la0), "v"(ga0), "v"(la1), "v"(ga1),
              "v"(lb0), "v"(gb0), "v"(lb1), "v"(gb1)
            : "memory");
    };

    uint4 ra[2], rb[2];
    auto loadEdge = [&](int kk) {
        ra[0] = (m0 + r0 < M) ? *(const uint4*)(A + (size_t)(m0 + r0) * K + kk + k0e)
                              : make_uint4(0u, 0u, 0u, 0u);
        ra[1] = (m0 + r1 < M) ? *(const uint4*)(A + (size_t)(m0 + r1) * K + kk + k1e)
                              : make_uint4(0u, 0u, 0u, 0u);
        rb[0] = (n0 + r0 < N) ? *(const uint4*)(W + (size_t)(n0 + r0) * K + kk + k0e)
                              : make_uint4(0u, 0u, 0u, 0u);
        rb[1] = (n0 + r1 < N) ? *(const uint4*)(W + (size_t)(n0 + r1) * K + kk + k1e)
                              : make_uint4(0u, 0u, 0u, 0u);
    };
    auto storeEdge = [&](int buf) {
        *(uint4*)(sm + A_LOFF(buf) / 2 + r0 * BKK + k0e) = ra[0];
        *(uint4*)(sm + A_LOFF(buf) / 2 + r1 * BKK + k1e) = ra[1];
        *(uint4*)(sm + B_LOFF(buf) / 2 + r0 * BKK + k0e) = rb[0];
        *(uint4*)(sm + B_LOFF(buf) / 2 + r1 * BKK + k1e) = rb[1];
    };

    auto compute = [&](int buf) {
        const unsigned short* At = sm + A_LOFF(buf) / 2;
        const unsigned short* Bt = sm + B_LOFF(buf) / 2;
        Frag a[2], b[4];
#pragma unroll
        for (int ti = 0; ti < 2; ++ti) {
            // A-matrix 16x32 layout: lanes0-15 K={0..7,16..23},
            // lanes16-31 K={8..15,24..31}
            int r  = wm * 32 + ti * 16 + (lane & 15);
            int kf = (lane >> 4) * 8;
            a[ti].u[0] = *(const uint4*)(At + r * BKK + kf);
            a[ti].u[1] = *(const uint4*)(At + r * BKK + 16 + kf);
        }
#pragma unroll
        for (int tj = 0; tj < 4; ++tj) {
            // B-matrix 32x16: N = lane%16; lanes0-15 K={0..7,16..23}, lanes16-31 K={8..15,24..31}
            int n  = wn * 64 + tj * 16 + (lane & 15);
            int kb = (lane >> 4) * 8;
            b[tj].u[0] = *(const uint4*)(Bt + n * BKK + kb);
            b[tj].u[1] = *(const uint4*)(Bt + n * BKK + kb + 16);
        }
#pragma unroll
        for (int ti = 0; ti < 2; ++ti)
#pragma unroll
            for (int tj = 0; tj < 4; ++tj)
                acc[ti][tj] = __builtin_amdgcn_wmma_f32_16x16x32_bf16(
                    false, a[ti].v, false, b[tj].v,
                    (short)0, acc[ti][tj], false, false);
    };

    int buf = 0;
    if (interior) {
        issueAsync(0, 0);
        asm volatile("s_wait_asynccnt 0x0" ::: "memory");
        __syncthreads();
        for (int kk = BKK; kk < K; kk += BKK) {
            issueAsync(kk, buf ^ 1);     // async DMA into other buffer
            compute(buf);                // WMMA on current buffer
            asm volatile("s_wait_asynccnt 0x0" ::: "memory");
            __syncthreads();
            buf ^= 1;
        }
        compute(buf);
    } else {
        loadEdge(0);
        storeEdge(0);
        __syncthreads();
        for (int kk = BKK; kk < K; kk += BKK) {
            loadEdge(kk);
            compute(buf);
            __syncthreads();
            storeEdge(buf ^ 1);
            __syncthreads();
            buf ^= 1;
        }
        compute(buf);
    }

    // Epilogue: C/D layout — VGPR r holds M=r (lanes0-15) / M=r+8 (lanes16-31)
#pragma unroll
    for (int ti = 0; ti < 2; ++ti) {
#pragma unroll
        for (int tj = 0; tj < 4; ++tj) {
            int col  = n0 + wn * 64 + tj * 16 + (lane & 15);
            int rowb = m0 + wm * 32 + ti * 16 + ((lane >> 4) << 3);
#pragma unroll
            for (int r = 0; r < 8; ++r) {
                int row = rowb + r;
                if (row < M && col < N) {
                    float v = acc[ti][tj][r];
                    if (bias) v += bias[col];
                    if (relu) v = fmaxf(v, 0.0f);
                    C[(size_t)row * N + col] = v;
                }
            }
        }
    }
}

// ---------------------------------------------------------------------------
// f32 -> bf16 convert with column padding (zero-fills cols >= cols)
// ---------------------------------------------------------------------------
__global__ void cvt_pad_kernel(const float* __restrict__ src,
                               unsigned short* __restrict__ dst,
                               int rows, int cols, int colsPad)
{
    int total = rows * colsPad;
    for (int i = blockIdx.x * 256 + threadIdx.x; i < total;
         i += gridDim.x * 256) {
        int r = i / colsPad, c = i - r * colsPad;
        dst[i] = (c < cols) ? f2bf(src[(size_t)r * cols + c])
                            : (unsigned short)0;
    }
}

// slice columns [off, off+outCols) of f32 src (rows x srcCols) -> bf16
__global__ void slice_cvt_kernel(const float* __restrict__ src,
                                 unsigned short* __restrict__ dst,
                                 int rows, int srcCols, int off, int outCols)
{
    int total = rows * outCols;
    for (int i = blockIdx.x * 256 + threadIdx.x; i < total;
         i += gridDim.x * 256) {
        int r = i / outCols, c = i - r * outCols;
        dst[i] = f2bf(src[(size_t)r * srcCols + off + c]);
    }
}

__global__ void fill_kernel(float* dst, float v, int n)
{
    int i = blockIdx.x * 256 + threadIdx.x;
    if (i < n) dst[i] = v;
}

// ---------------------------------------------------------------------------
// Key-padding mask:  nz[b,s] = any(x[b,s] != 0);  pad[b,s] = !(any nz >= s),
// pad[b,0] forced false (reference semantics).
// ---------------------------------------------------------------------------
__global__ __launch_bounds__(256)
void nz_kernel(const float* __restrict__ x, int* __restrict__ nz)
{
    int bs = blockIdx.x;
    const float* p = x + (size_t)bs * FLAT_IN;
    int any = 0;
    for (int i = threadIdx.x; i < FLAT_IN; i += 256)
        any |= (p[i] != 0.0f);
    __shared__ int sh[256];
    sh[threadIdx.x] = any;
    __syncthreads();
    for (int off = 128; off > 0; off >>= 1) {
        if (threadIdx.x < off) sh[threadIdx.x] |= sh[threadIdx.x + off];
        __syncthreads();
    }
    if (threadIdx.x == 0) nz[bs] = sh[0];
}

__global__ void pad_suffix_kernel(const int* __restrict__ nz,
                                  int* __restrict__ pad)
{
    int b = blockIdx.x;
    if (threadIdx.x == 0) {
        int any = 0;
        for (int s = S_LEN - 1; s >= 0; --s) {
            any |= nz[b * S_LEN + s];
            pad[b * S_LEN + s] = !any;
        }
        pad[b * S_LEN] = 0;
    }
}

// ---------------------------------------------------------------------------
// (B,S,D) -> (S,B,D) with sinusoidal positional encoding added
// ---------------------------------------------------------------------------
__global__ void add_pe_kernel(const float* __restrict__ src,
                              float* __restrict__ dst)
{
    int total = S_LEN * BATCH * DMODEL;
    for (int i = blockIdx.x * 256 + threadIdx.x; i < total;
         i += gridDim.x * 256) {
        int d  = i & (DMODEL - 1);
        int sb = i >> 9;
        int b  = sb % BATCH;
        int s  = sb / BATCH;
        int j  = d >> 1;
        float div = __expf(-(float)(2 * j) * 0.017988946f); // ln(1e4)/512
        float pe  = (d & 1) ? __cosf((float)s * div) : __sinf((float)s * div);
        dst[i] = src[((size_t)b * S_LEN + s) * DMODEL + d] + pe;
    }
}

// ---------------------------------------------------------------------------
// LayerNorm(x + r) * g + b ; r may be null; out may alias x. D = 512.
// ---------------------------------------------------------------------------
__global__ __launch_bounds__(128)
void ln_kernel(const float* __restrict__ x, const float* __restrict__ r,
               const float* __restrict__ g, const float* __restrict__ bt,
               float* __restrict__ out)
{
    int row = blockIdx.x, t = threadIdx.x;
    __shared__ float sh[128];
    const float* xr = x + (size_t)row * DMODEL;
    const float* rr = r ? r + (size_t)row * DMODEL : nullptr;
    float v[4];
    float sum = 0.0f;
#pragma unroll
    for (int i = 0; i < 4; ++i) {
        float a = xr[t + i * 128] + (rr ? rr[t + i * 128] : 0.0f);
        v[i] = a; sum += a;
    }
    sh[t] = sum; __syncthreads();
    for (int off = 64; off > 0; off >>= 1) {
        if (t < off) sh[t] += sh[t + off];
        __syncthreads();
    }
    float mean = sh[0] * (1.0f / DMODEL);
    __syncthreads();
    float vs = 0.0f;
#pragma unroll
    for (int i = 0; i < 4; ++i) { float d = v[i] - mean; vs += d * d; }
    sh[t] = vs; __syncthreads();
    for (int off = 64; off > 0; off >>= 1) {
        if (t < off) sh[t] += sh[t + off];
        __syncthreads();
    }
    float inv = rsqrtf(sh[0] * (1.0f / DMODEL) + 1e-5f);
#pragma unroll
    for (int i = 0; i < 4; ++i) {
        int c = t + i * 128;
        out[(size_t)row * DMODEL + c] = (v[i] - mean) * inv * g[c] + bt[c];
    }
}

// ---------------------------------------------------------------------------
// Encoder self-attention, D=512 H=4 HD=128, seq-first (S,B,*) layout.
// qkv: (S*B, 1536) fused;  out: (S*B, 512).  Key-padding mask applied.
// ---------------------------------------------------------------------------
__global__ __launch_bounds__(128)
void enc_attn_kernel(const float* __restrict__ qkv,
                     const int* __restrict__ pad,
                     float* __restrict__ out)
{
    const int q = blockIdx.x, b = blockIdx.y, h = blockIdx.z;
    const int t = threadIdx.x;
    __shared__ float qs[128], sc[128], tmp[128];

    qs[t] = qkv[((size_t)(q * BATCH + b)) * 1536 + h * HDIM + t];
    __syncthreads();

    float s = -1e30f;
    if (t < S_LEN) {
        const float* kp = qkv + ((size_t)(t * BATCH + b)) * 1536 + DMODEL + h * HDIM;
        float a = 0.0f;
#pragma unroll 4
        for (int d = 0; d < HDIM; ++d) a += qs[d] * kp[d];
        s = a * 0.08838834764831845f;           // 1/sqrt(128)
        if (pad[b * S_LEN + t]) s = -1e9f;
    }
    tmp[t] = s; __syncthreads();
    for (int off = 64; off > 0; off >>= 1) {
        if (t < off) tmp[t] = fmaxf(tmp[t], tmp[t + off]);
        __syncthreads();
    }
    float mx = tmp[0]; __syncthreads();
    float e = (t < S_LEN) ? __expf(s - mx) : 0.0f;
    sc[t] = e; tmp[t] = e; __syncthreads();
    for (int off = 64; off > 0; off >>= 1) {
        if (t < off) tmp[t] += tmp[t + off];
        __syncthreads();
    }
    float inv = 1.0f / tmp[0];

    float o = 0.0f;
    for (int k = 0; k < S_LEN; ++k)
        o += sc[k] * inv *
             qkv[((size_t)(k * BATCH + b)) * 1536 + 2 * DMODEL + h * HDIM + t];
    out[((size_t)(q * BATCH + b)) * DMODEL + h * HDIM + t] = o;
}

// ---------------------------------------------------------------------------
// Decoder cross-attention: q from qt (B,1536 fused, q at offset 0);
// k,v from kv (S*B, 1024).  No mask.  out: (B, 512).
// ---------------------------------------------------------------------------
__global__ __launch_bounds__(128)
void ca_attn_kernel(const float* __restrict__ qt,
                    const float* __restrict__ kv,
                    float* __restrict__ out)
{
    const int b = blockIdx.x, h = blockIdx.y, t = threadIdx.x;
    __shared__ float qs[128], sc[128], tmp[128];

    qs[t] = qt[(size_t)b * 1536 + h * HDIM + t];
    __syncthreads();

    float s = -1e30f;
    if (t < S_LEN) {
        const float* kp = kv + (size_t)(t * BATCH + b) * 1024 + h * HDIM;
        float a = 0.0f;
#pragma unroll 4
        for (int d = 0; d < HDIM; ++d) a += qs[d] * kp[d];
        s = a * 0.08838834764831845f;
    }
    tmp[t] = s; __syncthreads();
    for (int off = 64; off > 0; off >>= 1) {
        if (t < off) tmp[t] = fmaxf(tmp[t], tmp[t + off]);
        __syncthreads();
    }
    float mx = tmp[0]; __syncthreads();
    float e = (t < S_LEN) ? __expf(s - mx) : 0.0f;
    sc[t] = e; tmp[t] = e; __syncthreads();
    for (int off = 64; off > 0; off >>= 1) {
        if (t < off) tmp[t] += tmp[t + off];
        __syncthreads();
    }
    float inv = 1.0f / tmp[0];

    float o = 0.0f;
    for (int k = 0; k < S_LEN; ++k)
        o += sc[k] * inv * kv[(size_t)(k * BATCH + b) * 1024 + DMODEL + h * HDIM + t];
    out[(size_t)b * DMODEL + h * HDIM + t] = o;
}

// ---------------------------------------------------------------------------
// Adaptive average pool (143,111) -> (135,103), PyTorch window semantics.
// ---------------------------------------------------------------------------
__global__ void pool_kernel(const float* __restrict__ src,
                            float* __restrict__ dst)
{
    int total = BATCH * HOUT_ * WOUT_;
    for (int i = blockIdx.x * 256 + threadIdx.x; i < total;
         i += gridDim.x * 256) {
        int ow = i % WOUT_;
        int rest = i / WOUT_;
        int oh = rest % HOUT_;
        int b  = rest / HOUT_;
        int rs = (oh * HIN_) / HOUT_;
        int re = ((oh + 1) * HIN_ + HOUT_ - 1) / HOUT_;
        int cs = (ow * WIN_) / WOUT_;
        int ce = ((ow + 1) * WIN_ + WOUT_ - 1) / WOUT_;
        float s = 0.0f;
        for (int rr = rs; rr < re; ++rr)
            for (int cc = cs; cc < ce; ++cc)
                s += src[(size_t)b * FLAT_IN + rr * WIN_ + cc];
        dst[i] = s / (float)((re - rs) * (ce - cs));
    }
}

// ---------------------------------------------------------------------------
// Host-side orchestration
// ---------------------------------------------------------------------------
static inline int cdiv(int a, int b) { return (a + b - 1) / b; }

static void launch_cvt(hipStream_t st, const float* src, unsigned short* dst,
                       int rows, int cols, int colsPad)
{
    int total = rows * colsPad;
    int blocks = cdiv(total, 256);
    cvt_pad_kernel<<<blocks, 256, 0, st>>>(src, dst, rows, cols, colsPad);
}

static void launch_gemm(hipStream_t st, const unsigned short* A,
                        const unsigned short* W, const float* bias, float* C,
                        int M, int N, int K, int relu)
{
    dim3 grid(cdiv(N, BN), cdiv(M, BM));
    gemm_bf16_kernel<<<grid, 256, GEMM_LDS_BYTES, st>>>(A, W, bias, C, M, N, K, relu);
}

extern "C" void kernel_launch(void* const* d_in, const int* in_sizes, int n_in,
                              void* d_out, int out_size, void* d_ws, size_t ws_size,
                              hipStream_t stream)
{
    (void)in_sizes; (void)n_in; (void)out_size; (void)ws_size;

    const float* x        = (const float*)d_in[0];
    const float* fc1_w    = (const float*)d_in[1];
    const float* fc1_b    = (const float*)d_in[2];
    const float* fc2_w    = (const float*)d_in[3];
    const float* fc2_b    = (const float*)d_in[4];
    const float* fc3_w    = (const float*)d_in[5];
    const float* fc3_b    = (const float*)d_in[6];
    const float* out1_w   = (const float*)d_in[7];
    const float* out1_b   = (const float*)d_in[8];
    const float* out2_w   = (const float*)d_in[9];
    const float* out2_b   = (const float*)d_in[10];
    const float* out3_w   = (const float*)d_in[11];
    const float* out3_b   = (const float*)d_in[12];
    const float* enc_qkv_w  = (const float*)d_in[13];
    const float* enc_qkv_b  = (const float*)d_in[14];
    const float* enc_proj_w = (const float*)d_in[15];
    const float* enc_proj_b = (const float*)d_in[16];
    const float* enc_ff1_w  = (const float*)d_in[17];
    const float* enc_ff1_b  = (const float*)d_in[18];
    const float* enc_ff2_w  = (const float*)d_in[19];
    const float* enc_ff2_b  = (const float*)d_in[20];
    const float* enc_ln1_g  = (const float*)d_in[21];
    const float* enc_ln1_b  = (const float*)d_in[22];
    const float* enc_ln2_g  = (const float*)d_in[23];
    const float* enc_ln2_b  = (const float*)d_in[24];
    const float* enc_lnf_g  = (const float*)d_in[25];
    const float* enc_lnf_b  = (const float*)d_in[26];
    const float* dec_sa_qkv_w  = (const float*)d_in[27];
    const float* dec_sa_qkv_b  = (const float*)d_in[28];
    const float* dec_sa_proj_w = (const float*)d_in[29];
    const float* dec_sa_proj_b = (const float*)d_in[30];
    const float* dec_ca_qkv_w  = (const float*)d_in[31];
    const float* dec_ca_qkv_b  = (const float*)d_in[32];
    const float* dec_ca_proj_w = (const float*)d_in[33];
    const float* dec_ca_proj_b = (const float*)d_in[34];
    const float* dec_ff1_w  = (const float*)d_in[35];
    const float* dec_ff1_b  = (const float*)d_in[36];
    const float* dec_ff2_w  = (const float*)d_in[37];
    const float* dec_ff2_b  = (const float*)d_in[38];
    const float* dec_ln1_g  = (const float*)d_in[39];
    const float* dec_ln1_b  = (const float*)d_in[40];
    const float* dec_ln2_g  = (const float*)d_in[41];
    const float* dec_ln2_b  = (const float*)d_in[42];
    const float* dec_ln3_g  = (const float*)d_in[43];
    const float* dec_ln3_b  = (const float*)d_in[44];
    const float* dec_lnf_g  = (const float*)d_in[45];
    const float* dec_lnf_b  = (const float*)d_in[46];

    const int BS = BATCH * S_LEN;     // 3200

    // ---- workspace carve ----------------------------------------------------
    char* wp = (char*)d_ws;
    auto carve = [&](size_t bytes) -> char* {
        char* r = wp;
        wp += (bytes + 255) & ~(size_t)255;
        return r;
    };
    unsigned short* W_BF = (unsigned short*)carve((size_t)4096 * FLAT_PAD * 2);
    unsigned short* A_BF = (unsigned short*)carve((size_t)BS * FLAT_PAD * 2);
    float* P0   = (float*)carve((size_t)BS * 4096 * 4);
    float* P1   = (float*)carve((size_t)BS * 4096 * 4);
    float* HB   = (float*)carve((size_t)BS * DMODEL * 4);   // running residual (S,B,D)
    float* MEMF = (float*)carve((size_t)BS * DMODEL * 4);   // encoder memory
    float* KV   = (float*)carve((size_t)BS * 1024 * 4);     // decoder cross k,v
    float* TR   = (float*)carve((size_t)BATCH * DMODEL * 4);// decoder residual
    float* QT   = (float*)carve((size_t)BATCH * 1536 * 4);  // fused qkv (decoder)
    float* CAO  = (float*)carve((size_t)BATCH * DMODEL * 4);
    float* PO   = (float*)carve((size_t)BATCH * DMODEL * 4);
    float* FFB  = (float*)carve((size_t)BATCH * 1024 * 4);
    float* O1   = (float*)carve((size_t)BATCH * 2048 * 4);
    int*   NZ   = (int*)carve((size_t)BS * 4);
    int*   PADM = (int*)carve((size_t)BS * 4);

    // ---- padding mask -------------------------------------------------------
    nz_kernel<<<BS, 256, 0, stream>>>(x, NZ);
    pad_suffix_kernel<<<BATCH, 32, 0, stream>>>(NZ, PADM);

    // ---- input MLP: 15873 -> 4096 -> 2048 -> 512 (relu) ---------------------
    launch_cvt(stream, x, A_BF, BS, FLAT_IN, FLAT_PAD);
    launch_cvt(stream, fc1_w, W_BF, 4096, FLAT_IN, FLAT_PAD);
    launch_gemm(stream, A_BF, W_BF, fc1_b, P0, BS, 4096, FLAT_PAD, 1);

    launch_cvt(stream, P0, A_BF, BS, 4096, 4096);
    launch_cvt(stream, fc2_w, W_BF, 2048, 4096, 4096);
    launch_gemm(stream, A_BF, W_BF, fc2_b, P1, BS, 2048, 4096, 1);

    launch_cvt(stream, P1, A_BF, BS, 2048, 2048);
    launch_cvt(stream, fc3_w, W_BF, DMODEL, 2048, 2048);
    launch_gemm(stream, A_BF, W_BF, fc3_b, P0, BS, DMODEL, 2048, 1);

    // (B,S,D) -> (S,B,D) + positional encoding
    add_pe_kernel<<<cdiv(BS * DMODEL, 256), 256, 0, stream>>>(P0, HB);

    // ---- encoder: 2 post-norm layers ---------------------------------------
    for (int i = 0; i < 2; ++i) {
        const float* qkv_w = enc_qkv_w + (size_t)i * 1536 * DMODEL;
        const float* qkv_b = enc_qkv_b + (size_t)i * 1536;
        // fused QKV projection
        launch_cvt(stream, HB, A_BF, BS, DMODEL, DMODEL);
        launch_cvt(stream, qkv_w, W_BF, 1536, DMODEL, DMODEL);
        launch_gemm(stream, A_BF, W_BF, qkv_b, P0, BS, 1536, DMODEL, 0);
        // attention -> P1
        enc_attn_kernel<<<dim3(S_LEN, BATCH, NHEAD), 128, 0, stream>>>(P0, PADM, P1);
        // output projection
        launch_cvt(stream, P1, A_BF, BS, DMODEL, DMODEL);
        launch_cvt(stream, enc_proj_w + (size_t)i * DMODEL * DMODEL, W_BF,
                   DMODEL, DMODEL, DMODEL);
        launch_gemm(stream, A_BF, W_BF, enc_proj_b + (size_t)i * DMODEL, P0,
                    BS, DMODEL, DMODEL, 0);
        // residual + LN1
        ln_kernel<<<BS, 128, 0, stream>>>(HB, P0,
            enc_ln1_g + (size_t)i * DMODEL, enc_ln1_b + (size_t)i * DMODEL, HB);
        // FFN 512 -> 1024 (relu) -> 512
        launch_cvt(stream, HB, A_BF, BS, DMODEL, DMODEL);
        launch_cvt(stream, enc_ff1_w + (size_t)i * 1024 * DMODEL, W_BF,
                   1024, DMODEL, DMODEL);
        launch_gemm(stream, A_BF, W_BF, enc_ff1_b + (size_t)i * 1024, P0,
                    BS, 1024, DMODEL, 1);
        launch_cvt(stream, P0, A_BF, BS, 1024, 1024);
        launch_cvt(stream, enc_ff2_w + (size_t)i * DMODEL * 1024, W_BF,
                   DMODEL, 1024, 1024);
        launch_gemm(stream, A_BF, W_BF, enc_ff2_b + (size_t)i * DMODEL, P1,
                    BS, DMODEL, 1024, 0);
        // residual + LN2
        ln_kernel<<<BS, 128, 0, stream>>>(HB, P1,
            enc_ln2_g + (size_t)i * DMODEL, enc_ln2_b + (size_t)i * DMODEL, HB);
    }
    // final encoder norm -> memory
    ln_kernel<<<BS, 128, 0, stream>>>(HB, nullptr, enc_lnf_g, enc_lnf_b, MEMF);

    // ---- decoder: target = ones(1,B,D); 2 post-norm layers ------------------
    fill_kernel<<<cdiv(BATCH * DMODEL, 256), 256, 0, stream>>>(TR, 1.0f,
                                                               BATCH * DMODEL);
    for (int i = 0; i < 2; ++i) {
        // self-attention with Sq=Sk=1: softmax over 1 key -> output == v
        launch_cvt(stream, TR, A_BF, BATCH, DMODEL, DMODEL);
        launch_cvt(stream, dec_sa_qkv_w + (size_t)i * 1536 * DMODEL, W_BF,
                   1536, DMODEL, DMODEL);
        launch_gemm(stream, A_BF, W_BF, dec_sa_qkv_b + (size_t)i * 1536, QT,
                    BATCH, 1536, DMODEL, 0);
        // extract v (cols 1024..1535) as attention output, project
        slice_cvt_kernel<<<cdiv(BATCH * DMODEL, 256), 256, 0, stream>>>(
            QT, A_BF, BATCH, 1536, 2 * DMODEL, DMODEL);
        launch_cvt(stream, dec_sa_proj_w + (size_t)i * DMODEL * DMODEL, W_BF,
                   DMODEL, DMODEL, DMODEL);
        launch_gemm(stream, A_BF, W_BF, dec_sa_proj_b + (size_t)i * DMODEL, PO,
                    BATCH, DMODEL, DMODEL, 0);
        ln_kernel<<<BATCH, 128, 0, stream>>>(TR, PO,
            dec_ln1_g + (size_t)i * DMODEL, dec_ln1_b + (size_t)i * DMODEL, TR);

        // cross-attention: q from t, k/v from memory
        launch_cvt(stream, TR, A_BF, BATCH, DMODEL, DMODEL);
        launch_cvt(stream, dec_ca_qkv_w + (size_t)i * 1536 * DMODEL, W_BF,
                   1536, DMODEL, DMODEL);
        launch_gemm(stream, A_BF, W_BF, dec_ca_qkv_b + (size_t)i * 1536, QT,
                    BATCH, 1536, DMODEL, 0);
        // k,v projection of memory: weight rows [D, 3D)
        launch_cvt(stream, MEMF, A_BF, BS, DMODEL, DMODEL);
        launch_cvt(stream,
                   dec_ca_qkv_w + (size_t)i * 1536 * DMODEL + (size_t)DMODEL * DMODEL,
                   W_BF, 1024, DMODEL, DMODEL);
        launch_gemm(stream, A_BF, W_BF,
                    dec_ca_qkv_b + (size_t)i * 1536 + DMODEL, KV,
                    BS, 1024, DMODEL, 0);
        ca_attn_kernel<<<dim3(BATCH, NHEAD), 128, 0, stream>>>(QT, KV, CAO);
        launch_cvt(stream, CAO, A_BF, BATCH, DMODEL, DMODEL);
        launch_cvt(stream, dec_ca_proj_w + (size_t)i * DMODEL * DMODEL, W_BF,
                   DMODEL, DMODEL, DMODEL);
        launch_gemm(stream, A_BF, W_BF, dec_ca_proj_b + (size_t)i * DMODEL, PO,
                    BATCH, DMODEL, DMODEL, 0);
        ln_kernel<<<BATCH, 128, 0, stream>>>(TR, PO,
            dec_ln2_g + (size_t)i * DMODEL, dec_ln2_b + (size_t)i * DMODEL, TR);

        // FFN
        launch_cvt(stream, TR, A_BF, BATCH, DMODEL, DMODEL);
        launch_cvt(stream, dec_ff1_w + (size_t)i * 1024 * DMODEL, W_BF,
                   1024, DMODEL, DMODEL);
        launch_gemm(stream, A_BF, W_BF, dec_ff1_b + (size_t)i * 1024, FFB,
                    BATCH, 1024, DMODEL, 1);
        launch_cvt(stream, FFB, A_BF, BATCH, 1024, 1024);
        launch_cvt(stream, dec_ff2_w + (size_t)i * DMODEL * 1024, W_BF,
                   DMODEL, 1024, 1024);
        launch_gemm(stream, A_BF, W_BF, dec_ff2_b + (size_t)i * DMODEL, PO,
                    BATCH, DMODEL, 1024, 0);
        ln_kernel<<<BATCH, 128, 0, stream>>>(TR, PO,
            dec_ln3_g + (size_t)i * DMODEL, dec_ln3_b + (size_t)i * DMODEL, TR);
    }
    ln_kernel<<<BATCH, 128, 0, stream>>>(TR, nullptr, dec_lnf_g, dec_lnf_b, TR);

    // ---- output MLP: 512 -> 2048 -> 4096 -> 15873 (all relu) ---------------
    launch_cvt(stream, TR, A_BF, BATCH, DMODEL, DMODEL);
    launch_cvt(stream, out1_w, W_BF, 2048, DMODEL, DMODEL);
    launch_gemm(stream, A_BF, W_BF, out1_b, O1, BATCH, 2048, DMODEL, 1);

    launch_cvt(stream, O1, A_BF, BATCH, 2048, 2048);
    launch_cvt(stream, out2_w, W_BF, 4096, 2048, 2048);
    launch_gemm(stream, A_BF, W_BF, out2_b, P0, BATCH, 4096, 2048, 1);

    launch_cvt(stream, P0, A_BF, BATCH, 4096, 4096);
    launch_cvt(stream, out3_w, W_BF, FLAT_IN, 4096, 4096);
    launch_gemm(stream, A_BF, W_BF, out3_b, P1, BATCH, FLAT_IN, 4096, 1);

    // ---- adaptive average pool -> d_out ------------------------------------
    pool_kernel<<<cdiv(BATCH * HOUT_ * WOUT_, 256), 256, 0, stream>>>(
        P1, (float*)d_out);
}
